// GGNN_22222160789777
// MI455X (gfx1250) — matrix operand order repaired and checked by reading, not compiled, not run on hardware
//
#include <hip/hip_runtime.h>

typedef __attribute__((ext_vector_type(16))) _Float16 v16h;
typedef __attribute__((ext_vector_type(8)))  _Float16 v8h;
typedef __attribute__((ext_vector_type(8)))  float    v8f;

#define NN   70
#define EE   3
#define SS   10
#define NP   80      // padded nodes (5 M-tiles)
#define KP   256     // padded aggregation K: 3 regions of 80, padded to 256 (8 K-chunks)
#define KCH  8
#define WAVES 4
#define BLK   128

// ---- shared (per-workgroup) LDS offsets (bytes) ----
#define OFF_A     0                       // 80*256 f16 = 40960
#define OFF_WIN   40960                   // 3*512 f16 (fragment order) = 3072
#define OFF_WR    (OFF_WIN + 3072)        // 512 f16 = 1024
#define OFF_WZ    (OFF_WR + 1024)
#define OFF_WH    (OFF_WZ + 1024)
#define OFF_WO1   (OFF_WH + 1024)
#define OFF_BIN   (OFF_WO1 + 1024)        // 3*16 f32 = 192
#define OFF_BR    (OFF_BIN + 192)
#define OFF_BZ    (OFF_BR + 64)
#define OFF_BH    (OFF_BZ + 64)
#define OFF_BO1   (OFF_BH + 64)
#define OFF_WO2   (OFF_BO1 + 64)
#define SH_BYTES  (OFF_WO2 + 64)          // 48640

// ---- per-wave LDS offsets (bytes) ----
#define WOFF_PROPH  0                     // [80][32] f16 = 5120
#define WOFF_PROP32 5120                  // [80][16] f32 = 5120
#define WOFF_INST   10240                 // insT [16][256] f16 = 8192
#define WOFF_CATH   18432                 // [80][32] f16 = 5120
#define WOFF_Z32    23552                 // [80][16] f32 = 5120
#define WOFF_ANN    28672                 // 80 f32 = 320
#define PW_BYTES    28992
#define SMEM_TOTAL  (SH_BYTES + WAVES * PW_BYTES)   // 164608

__device__ __forceinline__ float sigm(float x) { return 1.0f / (1.0f + __expf(-x)); }

// A-matrix fragment (16x32 f16) from row-major [m][ld] LDS; contiguous k order (k = hi*16 + s):
// lane reads two contiguous 16B chunks: row[hi*16 .. +7] and row[hi*16+8 .. +7].
__device__ __forceinline__ v16h frag_a(const _Float16* p, int ld, int mb, int kb, int lane) {
  const int hi = lane >> 4;
  const _Float16* row = p + (mb + (lane & 15)) * ld + kb + hi * 16;
  v8h lo = *(const v8h*)row;
  v8h up = *(const v8h*)(row + 8);
  return __builtin_shufflevector(lo, up, 0,1,2,3,4,5,6,7,8,9,10,11,12,13,14,15);
}

// Pre-swizzled B fragment: lane's 16 halves stored contiguously (32B aligned).
__device__ __forceinline__ v16h frag_b_pre(const _Float16* F, int lane) {
  return *(const v16h*)(F + lane * 16);
}

// B fragment from transposed ins: lane (hi,n) needs k = kc*32 + hi*16 + {0..15},
// contiguous in insT[n*KP + k].
__device__ __forceinline__ v16h frag_b_insT(const _Float16* insT, int kc, int lane) {
  const _Float16* p = insT + (lane & 15) * KP + kc * 32 + (lane >> 4) * 16;
  return *(const v16h*)p;
}

// Pack 8 accumulator rows (+bias) into one 16B LDS store.
__device__ __forceinline__ void store8h(_Float16* p, const v8f& c, float bias) {
  v8h h;
#pragma unroll
  for (int v = 0; v < 8; ++v) h[v] = (_Float16)(c[v] + bias);
  *(v8h*)p = h;
}

__global__ __launch_bounds__(BLK) void ggnn_wmma_kernel(
    const float* __restrict__ ann_g, const float* __restrict__ A_g,
    const float* __restrict__ Win_g, const float* __restrict__ bin_g,
    const float* __restrict__ Wr_g,  const float* __restrict__ br_g,
    const float* __restrict__ Wz_g,  const float* __restrict__ bz_g,
    const float* __restrict__ Wh_g,  const float* __restrict__ bh_g,
    const float* __restrict__ Wo1_g, const float* __restrict__ bo1_g,
    const float* __restrict__ Wo2_g, const float* __restrict__ bo2_g,
    float* __restrict__ out_g) {
  extern __shared__ char smem[];
  _Float16* A_h  = (_Float16*)(smem + OFF_A);
  _Float16* WinF = (_Float16*)(smem + OFF_WIN);
  _Float16* WrF  = (_Float16*)(smem + OFF_WR);
  _Float16* WzF  = (_Float16*)(smem + OFF_WZ);
  _Float16* WhF  = (_Float16*)(smem + OFF_WH);
  _Float16* Wo1F = (_Float16*)(smem + OFF_WO1);
  float* binS = (float*)(smem + OFF_BIN);
  float* brS  = (float*)(smem + OFF_BR);
  float* bzS  = (float*)(smem + OFF_BZ);
  float* bhS  = (float*)(smem + OFF_BH);
  float* bo1S = (float*)(smem + OFF_BO1);
  float* Wo2S = (float*)(smem + OFF_WO2);

  const int t = threadIdx.x;
  const int lane = t & 31;
  const int wave = t >> 5;
  const int b = blockIdx.x * WAVES + wave;

  __builtin_prefetch(ann_g + (size_t)b * NN, 0, 0);

  // ---- A staged f16 with per-edge regions padded to 80 rows, zero elsewhere ----
  for (int i = t; i < NP * KP; i += BLK) {
    int m = i >> 8, k = i & 255;
    int e = k / 80, r = k - e * 80;
    bool ok = (m < NN) && (k < 240) && (r < NN);
    A_h[i] = ok ? (_Float16)A_g[m * (EE * NN) + e * NN + r] : (_Float16)0.f;
  }
  // ---- weight B-matrices staged directly in per-lane fragment order ----
  // frag element: lane l, slot j -> B[k=(l>>4)*16+j][n=l&15]
  for (int i = t; i < EE * 512; i += BLK) {
    int e = i >> 9, idx = i & 511, l = idx >> 4, j = idx & 15;
    int k = (l >> 4) * 16 + j, n = l & 15;
    WinF[i] = (k < SS && n < SS) ? (_Float16)Win_g[e * SS * SS + n * SS + k] : (_Float16)0.f;
  }
  for (int i = t; i < 512; i += BLK) {
    int l = i >> 4, j = i & 15;
    int k = (l >> 4) * 16 + j, n = l & 15;
    bool ok = (k < 2 * SS && n < SS);
    WrF[i]  = ok ? (_Float16)Wr_g[n * 2 * SS + k] : (_Float16)0.f;
    WzF[i]  = ok ? (_Float16)Wz_g[n * 2 * SS + k] : (_Float16)0.f;
    WhF[i]  = ok ? (_Float16)Wh_g[n * 2 * SS + k] : (_Float16)0.f;
    Wo1F[i] = (k < SS + 1 && n < SS) ? (_Float16)Wo1_g[n * (SS + 1) + k] : (_Float16)0.f;
  }
  if (t < 16) {
    brS[t]  = (t < SS) ? br_g[t]  : 0.f;
    bzS[t]  = (t < SS) ? bz_g[t]  : 0.f;
    bhS[t]  = (t < SS) ? bh_g[t]  : 0.f;
    bo1S[t] = (t < SS) ? bo1_g[t] : 0.f;
    Wo2S[t] = (t < SS) ? Wo2_g[t] : 0.f;
  }
  if (t < EE * 16) {
    int e = t >> 4, n = t & 15;
    binS[t] = (n < SS) ? bin_g[e * SS + n] : 0.f;
  }
  __syncthreads();

  char* wb = smem + SH_BYTES + wave * PW_BYTES;
  _Float16* propH  = (_Float16*)(wb + WOFF_PROPH);   // [80][32]
  float*    prop32 = (float*)(wb + WOFF_PROP32);     // [80][16]
  _Float16* insT   = (_Float16*)(wb + WOFF_INST);    // [16][256] transposed ins
  _Float16* catH   = (_Float16*)(wb + WOFF_CATH);    // [80][32]
  float*    z32    = (float*)(wb + WOFF_Z32);        // [80][16]
  float*    annF   = (float*)(wb + WOFF_ANN);        // [80]

  for (int m = lane; m < NP; m += 32) annF[m] = (m < NN) ? ann_g[(size_t)b * NN + m] : 0.f;
  __builtin_amdgcn_wave_barrier();
  for (int i = lane; i < NP * 32; i += 32) {
    int m = i >> 5, n = i & 31;
    float v = (n == 0) ? annF[m] : 0.f;
    propH[i] = (_Float16)v;
    if (n < 16) prop32[m * 16 + n] = v;
    if (n >= 20) catH[i] = (_Float16)0.f;
  }
  // zero insT tail (k = 240..255) once; never written again
  for (int i = lane; i < 16 * 16; i += 32) {
    int n = i >> 4, k = 240 + (i & 15);
    insT[n * KP + k] = (_Float16)0.f;
  }
  __builtin_amdgcn_wave_barrier();

  const int hi = lane >> 4;
  const int cn = lane & 15;

  for (int step = 0; step < 5; ++step) {
    // ---- ins[e] = prop @ W_in[e]^T + b_in[e], stored transposed (15 WMMA) ----
    {
      v16h bw0 = frag_b_pre(WinF, lane);
      v16h bw1 = frag_b_pre(WinF + 512, lane);
      v16h bw2 = frag_b_pre(WinF + 1024, lane);
      float bi0 = binS[cn], bi1 = binS[16 + cn], bi2 = binS[32 + cn];
      for (int mt = 0; mt < 5; ++mt) {
        v16h ap = frag_a(propH, 32, mt * 16, 0, lane);
        _Float16* dst = insT + cn * KP + mt * 16 + hi * 8;
        v8f c0 = {}, c1 = {}, c2 = {};
        c0 = __builtin_amdgcn_wmma_f32_16x16x32_f16(false, ap, false, bw0, (short)0, c0, false, false);
        c1 = __builtin_amdgcn_wmma_f32_16x16x32_f16(false, ap, false, bw1, (short)0, c1, false, false);
        c2 = __builtin_amdgcn_wmma_f32_16x16x32_f16(false, ap, false, bw2, (short)0, c2, false, false);
        store8h(dst, c0, bi0);
        store8h(dst + 80, c1, bi1);
        store8h(dst + 160, c2, bi2);
      }
    }
    __builtin_amdgcn_wave_barrier();

    // ---- a_in = A @ ins  (40 WMMA, reg-accumulated over 8 K-chunks) ----
    for (int mt = 0; mt < 5; ++mt) {
      v8f c = {};
      for (int kc = 0; kc < KCH; ++kc) {
        v16h aa = frag_a(A_h, KP, mt * 16, kc * 32, lane);
        v16h bi = frag_b_insT(insT, kc, lane);
        c = __builtin_amdgcn_wmma_f32_16x16x32_f16(false, aa, false, bi, (short)0, c, false, false);
      }
#pragma unroll
      for (int v = 0; v < 8; ++v) {
        int m = mt * 16 + v + hi * 8;
        if (cn < SS) catH[m * 32 + cn] = (_Float16)c[v];   // cols 0..9 = a_in
      }
    }
    // cols 10..19 = prop
    for (int i = lane; i < NP * SS; i += 32) {
      int m = i / SS, n = i - m * SS;
      catH[m * 32 + 10 + n] = propH[m * 32 + n];
    }
    __builtin_amdgcn_wave_barrier();

    // ---- z & r from one shared fragment per tile (10 WMMA) ----
    {
      v16h bwz = frag_b_pre(WzF, lane);
      v16h bwr = frag_b_pre(WrF, lane);
      for (int mt = 0; mt < 5; ++mt) {
        v16h ac = frag_a(catH, 32, mt * 16, 0, lane);
        v8f cz = {}, cr = {};
        cz = __builtin_amdgcn_wmma_f32_16x16x32_f16(false, ac, false, bwz, (short)0, cz, false, false);
        cr = __builtin_amdgcn_wmma_f32_16x16x32_f16(false, ac, false, bwr, (short)0, cr, false, false);
#pragma unroll
        for (int v = 0; v < 8; ++v) {
          int m = mt * 16 + v + hi * 8;
          z32[m * 16 + cn] = sigm(cz[v] + bzS[cn]);
          if (cn < SS) {
            float r = sigm(cr[v] + brS[cn]);
            catH[m * 32 + 10 + cn] = (_Float16)(r * prop32[m * 16 + cn]);
          }
        }
      }
    }
    __builtin_amdgcn_wave_barrier();

    // ---- h_hat = tanh([a_in, r*prop] @ Wh^T + bh); prop update (5 WMMA) ----
    {
      v16h bwh = frag_b_pre(WhF, lane);
      for (int mt = 0; mt < 5; ++mt) {
        v16h ac = frag_a(catH, 32, mt * 16, 0, lane);
        v8f c = {};
        c = __builtin_amdgcn_wmma_f32_16x16x32_f16(false, ac, false, bwh, (short)0, c, false, false);
#pragma unroll
        for (int v = 0; v < 8; ++v) {
          int m = mt * 16 + v + hi * 8;
          float h = tanhf(c[v] + bhS[cn]);
          float z = z32[m * 16 + cn];
          float p = prop32[m * 16 + cn];
          float np = (1.f - z) * p + z * h;
          prop32[m * 16 + cn] = np;
          propH[m * 32 + cn] = (_Float16)np;
        }
      }
    }
    __builtin_amdgcn_wave_barrier();
  }

  // ---- output head: tanh([prop, ann] @ Wo1^T + bo1) @ Wo2^T + bo2 ----
  for (int i = lane; i < NP * 32; i += 32) {
    int m = i >> 5, n = i & 31;
    _Float16 v = (_Float16)0.f;
    if (n < SS) v = propH[m * 32 + n];
    else if (n == SS) v = (_Float16)annF[m];
    catH[i] = v;
  }
  __builtin_amdgcn_wave_barrier();
  {
    v16h bw = frag_b_pre(Wo1F, lane);
    for (int mt = 0; mt < 5; ++mt) {
      v16h ac = frag_a(catH, 32, mt * 16, 0, lane);
      v8f c = {};
      c = __builtin_amdgcn_wmma_f32_16x16x32_f16(false, ac, false, bw, (short)0, c, false, false);
#pragma unroll
      for (int v = 0; v < 8; ++v) {
        int m = mt * 16 + v + hi * 8;
        z32[m * 16 + cn] = tanhf(c[v] + bo1S[cn]);
      }
    }
  }
  __builtin_amdgcn_wave_barrier();
  const float bo2v = bo2_g[0];
  for (int m = lane; m < NN; m += 32) {
    float s = bo2v;
#pragma unroll
    for (int j = 0; j < SS; ++j) s += z32[m * 16 + j] * Wo2S[j];
    out_g[(size_t)b * NN + m] = s;
  }
}

extern "C" void kernel_launch(void* const* d_in, const int* in_sizes, int n_in,
                              void* d_out, int out_size, void* d_ws, size_t ws_size,
                              hipStream_t stream) {
  (void)in_sizes; (void)n_in; (void)d_ws; (void)ws_size; (void)out_size;
  const float* ann = (const float*)d_in[0];
  const float* A   = (const float*)d_in[1];
  const float* Win = (const float*)d_in[2];
  const float* bin = (const float*)d_in[3];
  const float* Wr  = (const float*)d_in[4];
  const float* br  = (const float*)d_in[5];
  const float* Wz  = (const float*)d_in[6];
  const float* bz  = (const float*)d_in[7];
  const float* Wh  = (const float*)d_in[8];
  const float* bh  = (const float*)d_in[9];
  const float* Wo1 = (const float*)d_in[10];
  const float* bo1 = (const float*)d_in[11];
  const float* Wo2 = (const float*)d_in[12];
  const float* bo2 = (const float*)d_in[13];
  float* out = (float*)d_out;
  const int B = 16384;
  dim3 grid(B / WAVES), block(BLK);
  hipLaunchKernelGGL(ggnn_wmma_kernel, grid, block, SMEM_TOTAL, stream,
                     ann, A, Win, bin, Wr, br, Wz, bz, Wh, bh, Wo1, bo1, Wo2, bo2, out);
}
